// TokenLearner_24068996727090
// MI455X (gfx1250) — matrix operand order repaired and checked by reading, not compiled, not run on hardware
//
#include <hip/hip_runtime.h>
#include <hip/hip_bf16.h>

typedef __attribute__((ext_vector_type(16))) __bf16 v16bf;
typedef __attribute__((ext_vector_type(8)))  float  v8f;

namespace {

constexpr int   kDim   = 768;
constexpr int   kHW    = 1024;
constexpr int   kInner = 96;
constexpr int   kNTok  = 8;
constexpr float kEps   = 1e-5f;

constexpr int kTB  = 128;           // tokens per staged block
constexpr int kAP  = 776;           // atile row pitch in bf16 (1552B, 16B-aligned)
constexpr int kHP  = 97;            // hbuf row pitch in f32 (odd -> conflict-free)
constexpr int kKIt = kDim / 32;     // 24 K-steps for xn@w1
constexpr int kNT  = kInner / 16;   // 6 N-tiles

// LDS layout (bytes)
constexpr int MU_OFF  = 0;                          // mu[1024] f32
constexpr int RS_OFF  = MU_OFF + kHW * 4;           // rstd[1024] f32
constexpr int LG_OFF  = RS_OFF + kHW * 4;           // logits/attn [1024][8] f32
constexpr int RED_OFF = LG_OFF + kHW * kNTok * 4;   // red[256] + colmax[8] + colinv[8]
constexpr int W2_OFF  = RED_OFF + 272 * 4;          // w2 [96][8] f32
constexpr int H_OFF   = W2_OFF + kInner * kNTok * 4;// hbuf [8 waves][16][97] f32
constexpr int AT_OFF  = H_OFF + 8 * 16 * kHP * 4;   // atile [128][776] bf16
constexpr int LDS_BYTES = AT_OFF + kTB * kAP * 2;   // 293,440 B (< 320 KB)

union BF16x16 {
  v16bf v;
  unsigned short u[16];
  uint4 q[2];
};

__device__ inline unsigned short f2bf(float f) {
  union { float f; unsigned int u; } c;
  c.f = f;
  unsigned int u = c.u + 0x7fffu + ((c.u >> 16) & 1u);  // RNE to bf16
  return (unsigned short)(u >> 16);
}

__device__ inline unsigned int pack_bf2(float lo, float hi) {
  return (unsigned int)f2bf(lo) | ((unsigned int)f2bf(hi) << 16);
}

} // namespace

// Repack w1 [768,96] f32 -> bf16 WMMA B-fragment layout in d_ws.
// Index: [kk][nt][lane] * 8 dwords; lane<16: col n=nt*16+lane, K=kk*32+{0..15}
// packed 2/dword; lane>=16: K=kk*32+16+{0..15}.
__global__ void pack_w1_kernel(const float* __restrict__ w1,
                               unsigned int* __restrict__ out) {
  int idx = blockIdx.x * blockDim.x + threadIdx.x;
  if (idx >= kKIt * kNT * 32) return;
  int lane = idx & 31;
  int nt   = (idx >> 5) % kNT;
  int kk   = idx / (32 * kNT);
  int n    = nt * 16 + (lane & 15);
  int cb   = kk * 32 + (lane >> 4) * 16;
  unsigned int* dst = out + (size_t)idx * 8;
#pragma unroll
  for (int v = 0; v < 8; ++v) {
    float lo = w1[(size_t)(cb + 2 * v) * kInner + n];
    float hi = w1[(size_t)(cb + 2 * v + 1) * kInner + n];
    dst[v] = pack_bf2(lo, hi);
  }
}

__global__ void __launch_bounds__(256)
tokenlearner_kernel(const float* __restrict__ x,
                    const float* __restrict__ gamma,
                    const float* __restrict__ beta,
                    const float* __restrict__ b1,
                    const float* __restrict__ w2,
                    const float* __restrict__ b2,
                    const unsigned int* __restrict__ w1p,
                    float* __restrict__ out) {
  extern __shared__ char smem[];
  float* mu_s   = (float*)(smem + MU_OFF);
  float* rs_s   = (float*)(smem + RS_OFF);
  float* lg     = (float*)(smem + LG_OFF);
  float* red    = (float*)(smem + RED_OFF);
  float* colmax = red + 256;
  float* colinv = red + 264;
  float* w2_s   = (float*)(smem + W2_OFF);
  float* hbuf   = (float*)(smem + H_OFF);
  unsigned short* atile = (unsigned short*)(smem + AT_OFF);

  const int tid  = threadIdx.x;
  const int lane = tid & 31;
  const int wave = tid >> 5;
  const int b    = blockIdx.x;
  const float* xb = x + (size_t)b * kDim * kHW;

  // ---- Phase A: per-token mean / rstd over channels (pass 1 over x[b]) ----
  {
    for (int i = tid; i < kInner * kNTok; i += 256) w2_s[i] = w2[i];
    float s0=0.f,s1=0.f,s2=0.f,s3=0.f,q0=0.f,q1=0.f,q2=0.f,q3=0.f;
    for (int c = 0; c < kDim; ++c) {
      const float* row = xb + (size_t)c * kHW;
      float v0 = row[tid];
      float v1 = row[tid + 256];
      float v2 = row[tid + 512];
      float v3 = row[tid + 768];
      s0 += v0; q0 += v0 * v0;
      s1 += v1; q1 += v1 * v1;
      s2 += v2; q2 += v2 * v2;
      s3 += v3; q3 += v3 * v3;
    }
    const float invC = 1.0f / (float)kDim;
    float m;
    m = s0*invC; mu_s[tid      ] = m; rs_s[tid      ] = rsqrtf(q0*invC - m*m + kEps);
    m = s1*invC; mu_s[tid + 256] = m; rs_s[tid + 256] = rsqrtf(q1*invC - m*m + kEps);
    m = s2*invC; mu_s[tid + 512] = m; rs_s[tid + 512] = rsqrtf(q2*invC - m*m + kEps);
    m = s3*invC; mu_s[tid + 768] = m; rs_s[tid + 768] = rsqrtf(q3*invC - m*m + kEps);
  }
  __syncthreads();

  // ---- Phase B: per 128-token block: stage xn(bf16) -> LDS, WMMA h=gelu(xn@w1+b1),
  //               then logits = h@w2+b2 into LDS (pass 2 over x[b]) ----
  for (int blk = 0; blk < kHW / kTB; ++blk) {
    const int t0 = blk * kTB;
    for (int i = tid; i < kDim * kTB; i += 256) {
      int c  = i >> 7;
      int tl = i & (kTB - 1);
      float v = xb[(size_t)c * kHW + t0 + tl];
      int t = t0 + tl;
      float xn = (v - mu_s[t]) * rs_s[t] * gamma[c] + beta[c];
      atile[(size_t)tl * kAP + c] = f2bf(xn);
    }
    __syncthreads();

    v8f acc[kNT];
#pragma unroll
    for (int nt = 0; nt < kNT; ++nt)
      acc[nt] = (v8f){0.f,0.f,0.f,0.f,0.f,0.f,0.f,0.f};

    const unsigned short* arow = atile + (size_t)(wave * 16 + (lane & 15)) * kAP;
    const int khalf8 = (lane >> 4) * 8;
    for (int kk = 0; kk < kKIt; ++kk) {
      BF16x16 af;
      const int c0 = kk * 32;
      af.q[0] = *(const uint4*)(arow + c0 + 2 * khalf8);        // ds_load_b128
      af.q[1] = *(const uint4*)(arow + c0 + 2 * khalf8 + 8);    // ds_load_b128
#pragma unroll
      for (int nt = 0; nt < kNT; ++nt) {
        BF16x16 bf;
        const uint4* src =
            (const uint4*)(w1p + (size_t)((kk * kNT + nt) * 32 + lane) * 8);
        bf.q[0] = src[0];
        bf.q[1] = src[1];
        acc[nt] = __builtin_amdgcn_wmma_f32_16x16x32_bf16(
            false, af.v, false, bf.v, (short)0, acc[nt], false, false);
      }
    }

    // bias + exact GELU, spill h tile to LDS
#pragma unroll
    for (int nt = 0; nt < kNT; ++nt) {
      const int n  = nt * 16 + (lane & 15);
      const float bb = b1[n];
#pragma unroll
      for (int r = 0; r < 8; ++r) {
        int mrow = r + 8 * (lane >> 4);
        float v = acc[nt][r] + bb;
        float g = 0.5f * v * (1.0f + erff(v * 0.70710678118654752f));
        hbuf[(size_t)(wave * 16 + mrow) * kHP + n] = g;
      }
    }
    __syncthreads();

    // logits = h @ w2 + b2 (96x8, scalar FMAs from LDS)
    {
      const int tl = lane & 15;
      const float* hrow = hbuf + (size_t)(wave * 16 + tl) * kHP;
      const int t = t0 + wave * 16 + tl;
#pragma unroll
      for (int j4 = 0; j4 < 4; ++j4) {
        const int j = (lane >> 4) * 4 + j4;
        float s = b2[j];
        for (int n = 0; n < kInner; ++n) s += hrow[n] * w2_s[n * kNTok + j];
        lg[t * kNTok + j] = s;
      }
    }
    __syncthreads();
  }

  // ---- Softmax over 1024 tokens per slot; 1/sum folded into final store ----
  {
    const int j = tid & 7;
    const int grp = tid >> 3;
    float mx = -3.402823466e38f;
    for (int t = grp; t < kHW; t += 32) mx = fmaxf(mx, lg[t * kNTok + j]);
    red[tid] = mx;
    __syncthreads();
    if (tid < kNTok) {
      float m = red[tid];
      for (int g = 1; g < 32; ++g) m = fmaxf(m, red[g * kNTok + tid]);
      colmax[tid] = m;
    }
    __syncthreads();
    const float cm = colmax[j];
    float sm = 0.f;
    for (int t = grp; t < kHW; t += 32) {
      float e = __expf(lg[t * kNTok + j] - cm);
      lg[t * kNTok + j] = e;
      sm += e;
    }
    red[tid] = sm;
    __syncthreads();
    if (tid < kNTok) {
      float s = red[tid];
      for (int g = 1; g < 32; ++g) s += red[g * kNTok + tid];
      colinv[tid] = 1.0f / s;
    }
    __syncthreads();
  }

  // ---- Phase C: out[b,c,n] = (1/HW) * sum_t xn[t,c]*attn[t,n], WMMA K=1024
  //      (pass 3 over x[b]; xn recomputed on the fly, N padded 8->16) ----
  {
    const int mrow  = lane & 15;
    const int khalf = lane >> 4;
#pragma unroll 1
    for (int i = 0; i < 6; ++i) {
      const int c0 = (wave * 6 + i) * 16;
      const int cc = c0 + mrow;
      const float* px = xb + (size_t)cc * kHW;
      const float gam = gamma[cc];
      const float bet = beta[cc];
      v8f acc = (v8f){0.f,0.f,0.f,0.f,0.f,0.f,0.f,0.f};
      for (int kk = 0; kk < kHW / 32; ++kk) {
        const int tb = kk * 32;
        const int ta = tb + khalf * 8;
        const int tc = tb + 16 + khalf * 8;
        float va[8], vb2[8];
        *(float4*)&va[0]  = *(const float4*)(px + ta);      // global_load_b128
        *(float4*)&va[4]  = *(const float4*)(px + ta + 4);
        *(float4*)&vb2[0] = *(const float4*)(px + tc);
        *(float4*)&vb2[4] = *(const float4*)(px + tc + 4);
        BF16x16 af;
#pragma unroll
        for (int qi = 0; qi < 8; ++qi) {
          int t1 = ta + qi;
          int t2 = tc + qi;
          af.u[qi]     = f2bf((va[qi]  - mu_s[t1]) * rs_s[t1] * gam + bet);
          af.u[qi + 8] = f2bf((vb2[qi] - mu_s[t2]) * rs_s[t2] * gam + bet);
        }
        BF16x16 bf;
        const int n = lane & 15;
#pragma unroll
        for (int qi = 0; qi < 16; ++qi) {
          int t = tb + (qi < 8 ? khalf * 8 + qi : 16 + khalf * 8 + (qi - 8));
          float a = (n < kNTok) ? lg[t * kNTok + n] : 0.f;
          bf.u[qi] = f2bf(a);
        }
        acc = __builtin_amdgcn_wmma_f32_16x16x32_bf16(
            false, af.v, false, bf.v, (short)0, acc, false, false);
      }
      const int n = lane & 15;
      if (n < kNTok) {
        const float sc = colinv[n] * (1.0f / (float)kHW);
#pragma unroll
        for (int r = 0; r < 8; ++r) {
          int c = c0 + r + 8 * khalf;
          out[((size_t)b * kDim + c) * kNTok + n] = acc[r] * sc;
        }
      }
    }
  }
}

extern "C" void kernel_launch(void* const* d_in, const int* in_sizes, int n_in,
                              void* d_out, int out_size, void* d_ws, size_t ws_size,
                              hipStream_t stream) {
  (void)in_sizes; (void)n_in; (void)out_size; (void)ws_size;
  const float* x     = (const float*)d_in[0];
  const float* gamma = (const float*)d_in[1];
  const float* beta  = (const float*)d_in[2];
  const float* w1    = (const float*)d_in[3];
  const float* b1    = (const float*)d_in[4];
  const float* w2    = (const float*)d_in[5];
  const float* b2    = (const float*)d_in[6];
  float* out = (float*)d_out;

  // d_ws: bf16-fragment-packed w1 (24*6*32*8 dwords = 147,456 bytes)
  unsigned int* w1p = (unsigned int*)d_ws;

  pack_w1_kernel<<<(kKIt * kNT * 32 + 255) / 256, 256, 0, stream>>>(w1, w1p);
  tokenlearner_kernel<<<64, 256, LDS_BYTES, stream>>>(x, gamma, beta, b1, w2,
                                                      b2, w1p, out);
}
